// Mukara_29326036697604
// MI455X (gfx1250) — compile-verified, matrix operand order checked by reading, not run on hardware
//
#include <hip/hip_runtime.h>
#include <hip/hip_bf16.h>
#include <hip/hip_fp16.h>

// MI455X / gfx1250, wave32. Matrix op: v_wmma_f32_16x16x32_f16 (probe-confirmed builtin).
typedef __attribute__((ext_vector_type(16))) _Float16 v16h;
typedef __attribute__((ext_vector_type(8)))  float    v8f;

#define WMMA_F16(a, b, c) \
    __builtin_amdgcn_wmma_f32_16x16x32_f16(false, (a), false, (b), (short)0, (c), false, false)

// CDNA5 16-bit A/B operand layout for 16x16x32: lane dim = (M or N) = lane&15,
// VGPR v holds packed halves (k0, k0+1) with k0 = (v>>2)*16 + (lane>>4)*8 + (v&3)*2.
// Key observation: for fixed lane, dwords v=0..3 cover halves [h*8 .. h*8+7]
// (16 contiguous bytes) and v=4..7 cover [16+h*8 .. 16+h*8+7]  ->  b128 loads.
static __device__ __forceinline__ int a_k0(int v, int h) {
    return ((v >> 2) << 4) + (h << 3) + ((v & 3) << 1);
}

// Load a 16x32 f16 A/B-operand tile whose per-lane dwords are stored
// contiguously (32B per lane) at base + lane*8 dwords: 2x global_load_b128.
static __device__ __forceinline__ v16h load_packed_tile(const unsigned int* __restrict__ base,
                                                        int lane) {
    v16h b;
    uint4 lo = *(const uint4*)(base + lane * 8);
    uint4 hi = *(const uint4*)(base + lane * 8 + 4);
    unsigned int* u = (unsigned int*)&b;
    u[0] = lo.x; u[1] = lo.y; u[2] = lo.z; u[3] = lo.w;
    u[4] = hi.x; u[5] = hi.y; u[6] = hi.z; u[7] = hi.w;
    return b;
}

// Build an A tile from a row-major f16 row (32 halves): 2x b128 per lane.
static __device__ __forceinline__ v16h load_a_f16row(const _Float16* __restrict__ row, int h) {
    v16h a;
    uint4 lo = *(const uint4*)(row + h * 8);        // halves h*8 .. h*8+7   -> v=0..3
    uint4 hi = *(const uint4*)(row + 16 + h * 8);   // halves 16+h*8 .. +7   -> v=4..7
    unsigned int* u = (unsigned int*)&a;
    u[0] = lo.x; u[1] = lo.y; u[2] = lo.z; u[3] = lo.w;
    u[4] = hi.x; u[5] = hi.y; u[6] = hi.z; u[7] = hi.w;
    return a;
}

// Build an A tile from a row-major f32 row (32 floats): 4x b128 + cvt_pk.
static __device__ __forceinline__ v16h load_a_f32row(const float* __restrict__ row, int h) {
    v16h a;
    #pragma unroll
    for (int g = 0; g < 2; ++g) {
        float4 f0 = *(const float4*)(row + g * 16 + h * 8);
        float4 f1 = *(const float4*)(row + g * 16 + h * 8 + 4);
        a[g * 8 + 0] = (_Float16)f0.x; a[g * 8 + 1] = (_Float16)f0.y;
        a[g * 8 + 2] = (_Float16)f0.z; a[g * 8 + 3] = (_Float16)f0.w;
        a[g * 8 + 4] = (_Float16)f1.x; a[g * 8 + 5] = (_Float16)f1.y;
        a[g * 8 + 6] = (_Float16)f1.z; a[g * 8 + 7] = (_Float16)f1.w;
    }
    return a;
}

// ---------------------------------------------------------------------------
// Pack a row-major f32 weight matrix W[K][N] into f16 B-operand tiles.
// Tile order: tile = nt*KT + kt. Dword layout: dst[tile*256 + lane*8 + v]
// (per-lane contiguous 32B), so consumers fetch a tile with 2x b128 per lane.
// ---------------------------------------------------------------------------
__global__ __launch_bounds__(256) void pack_b_tiles(
    const float* __restrict__ W, int N, int KT, int NT,
    unsigned int* __restrict__ dst)
{
    int id = blockIdx.x * blockDim.x + threadIdx.x;
    int total = KT * NT * 256;
    if (id >= total) return;
    int v = id & 7, lane = (id >> 3) & 31, tile = id >> 8;
    int kt = tile % KT, nt = tile / KT;
    int n = nt * 16 + (lane & 15);
    int k = kt * 32 + a_k0(v, lane >> 4);
    union { _Float16 h[2]; unsigned int u; } cv;
    cv.h[0] = (_Float16)W[(long)k * N + n];
    cv.h[1] = (_Float16)W[(long)(k + 1) * N + n];
    dst[id] = cv.u;
}

// ---------------------------------------------------------------------------
// Node encoder: E = relu(X@W1+b1)@W2+b2, X:[nodes x 64], W1:[64x128], W2:[128x32].
// One wave per 16-row strip. blockIdx.y selects O (0) vs D (1) weight set.
// Output embeddings stored as f16 (halves gather traffic; feeds WMMA A directly).
// ---------------------------------------------------------------------------
#define NODE_WAVES 4
__global__ __launch_bounds__(128) void node_encode(
    const float* __restrict__ X,
    const unsigned int* __restrict__ packO1, const unsigned int* __restrict__ packO2,
    const float* __restrict__ bO1v, const float* __restrict__ bO2v, _Float16* __restrict__ dstO,
    const unsigned int* __restrict__ packD1, const unsigned int* __restrict__ packD2,
    const float* __restrict__ bD1v, const float* __restrict__ bD2v, _Float16* __restrict__ dstD,
    int nStrips)
{
    __shared__ __align__(16) _Float16 Hs[NODE_WAVES][16 * 128];
    const int wave = threadIdx.x >> 5, lane = threadIdx.x & 31;
    const int strip = blockIdx.x * NODE_WAVES + wave;
    if (strip >= nStrips) return;

    const unsigned int* p1 = blockIdx.y ? packD1 : packO1;
    const unsigned int* p2 = blockIdx.y ? packD2 : packO2;
    const float* b1 = blockIdx.y ? bD1v : bO1v;
    const float* b2 = blockIdx.y ? bD2v : bO2v;
    _Float16* dst = blockIdx.y ? dstD : dstO;
    const int m = lane & 15, h = lane >> 4;

    // A tiles (16x32 f16 x2) for this strip, f32 -> f16 (b128 loads + cvt_pk).
    const float* row = X + (long)(strip * 16 + m) * 64;
    v16h a0 = load_a_f32row(row, h);
    v16h a1 = load_a_f32row(row + 32, h);

    // Layer 1: 8 N-tiles, K = 64 (2 WMMA each). relu -> f16 -> LDS (row-major 16x128).
    #pragma unroll
    for (int nt = 0; nt < 8; ++nt) {
        v16h bt0 = load_packed_tile(p1 + (nt * 2 + 0) * 256, lane);
        v16h bt1 = load_packed_tile(p1 + (nt * 2 + 1) * 256, lane);
        v8f acc = {};
        acc = WMMA_F16(a0, bt0, acc);
        acc = WMMA_F16(a1, bt1, acc);
        int n = nt * 16 + m;
        float bb = b1[n];
        #pragma unroll
        for (int r = 0; r < 8; ++r) {
            float hv = fmaxf(acc[r] + bb, 0.f);
            Hs[wave][(r + h * 8) * 128 + n] = (_Float16)hv;   // C layout: M = r + h*8, N = lane&15
        }
    }
    // Data crosses lanes within the wave via LDS: fence DS ops (wave-private region,
    // so a wave-level s_wait_dscnt suffices; no workgroup barrier needed).
    __builtin_amdgcn_wave_barrier();
    asm volatile("s_wait_dscnt 0x0" ::: "memory");

    // Reload H as A tiles (K = 128 -> 4 tiles), 2x ds_load_b128 per tile.
    v16h ha[4];
    #pragma unroll
    for (int kt = 0; kt < 4; ++kt) {
        unsigned int* u = (unsigned int*)&ha[kt];
        #pragma unroll
        for (int g = 0; g < 2; ++g) {
            uint4 q = *(const uint4*)&Hs[wave][m * 128 + kt * 32 + g * 16 + h * 8];
            u[g * 4 + 0] = q.x; u[g * 4 + 1] = q.y; u[g * 4 + 2] = q.z; u[g * 4 + 3] = q.w;
        }
    }
    // Layer 2: 2 N-tiles, K = 128 (4 WMMA each). Store f16 embeddings.
    #pragma unroll
    for (int nt = 0; nt < 2; ++nt) {
        v8f acc = {};
        #pragma unroll
        for (int kt = 0; kt < 4; ++kt) {
            v16h bt = load_packed_tile(p2 + (nt * 4 + kt) * 256, lane);
            acc = WMMA_F16(ha[kt], bt, acc);
        }
        int n = nt * 16 + m;
        float bb = b2[n];
        #pragma unroll
        for (int r = 0; r < 8; ++r)
            dst[(long)(strip * 16 + r + h * 8) * 32 + n] = (_Float16)(acc[r] + bb);
    }
}

// ---------------------------------------------------------------------------
// Pair MLP: one wave per 16-pair strip. Gathers f16 embeddings into A-layout
// with 2x b128 per tile (4 gather requests per strip-lane total), 8 WMMAs per
// strip (K=64, N=64); t-column handled as rank-1 VALU update. Layer 2 (N=1)
// via shuffle reduction; softplus; per-wave deterministic partial sums.
// ---------------------------------------------------------------------------
#define PAIR_WAVES 8
__global__ __launch_bounds__(256) void pair_mlp(
    const _Float16* __restrict__ Oe, const _Float16* __restrict__ De,
    const int* __restrict__ Oi, const int* __restrict__ Di,
    const float* __restrict__ tOD,
    const unsigned int* __restrict__ packP1,
    const float* __restrict__ bP1, const float* __restrict__ wp1t,
    const float* __restrict__ WP2, const float* __restrict__ bP2,
    float* __restrict__ wave_sums, int nStrips)
{
    const int wave = threadIdx.x >> 5, lane = threadIdx.x & 31;
    const int wid = blockIdx.x * PAIR_WAVES + wave;
    const int nw = gridDim.x * PAIR_WAVES;
    const int m = lane & 15, h = lane >> 4;
    const float bP2v = bP2[0];

    // Hoist all 8 WP1 B-tiles into VGPRs (64 regs) -- reused every strip.
    v16h bO[4], bD[4];
    #pragma unroll
    for (int nt = 0; nt < 4; ++nt) {
        bO[nt] = load_packed_tile(packP1 + (nt * 2 + 0) * 256, lane);
        bD[nt] = load_packed_tile(packP1 + (nt * 2 + 1) * 256, lane);
    }
    float bb[4], wt[4], w2[4];
    #pragma unroll
    for (int nt = 0; nt < 4; ++nt) {
        int n = nt * 16 + m;
        bb[nt] = bP1[n]; wt[nt] = wp1t[n]; w2[nt] = WP2[n];
    }

    float wtotal = 0.f;
    for (int strip = wid; strip < nStrips; strip += nw) {
        int p = strip * 16 + m;
        int oi = Oi[p], di = Di[p];
        float tv = tOD[p];

        // L2-resident gathers (each table = 3.2MB f16): 2x b128 per operand.
        v16h aO = load_a_f16row(Oe + (long)oi * 32, h);
        v16h aD = load_a_f16row(De + (long)di * 32, h);

        float tm[8];
        #pragma unroll
        for (int r = 0; r < 8; ++r) tm[r] = __shfl(tv, r + h * 8, 32);

        float z[8] = {0, 0, 0, 0, 0, 0, 0, 0};
        #pragma unroll
        for (int nt = 0; nt < 4; ++nt) {
            v8f acc = {};
            acc = WMMA_F16(aO, bO[nt], acc);
            acc = WMMA_F16(aD, bD[nt], acc);
            #pragma unroll
            for (int r = 0; r < 8; ++r) {
                float hv = fmaxf(acc[r] + bb[nt] + tm[r] * wt[nt], 0.f);  // + t*WP1[64] + bias, relu
                z[r] += hv * w2[nt];                                      // layer-2 partial (N=1)
            }
        }
        // Reduce z over the 16 N-lanes of each half-wave.
        #pragma unroll
        for (int r = 0; r < 8; ++r) {
            z[r] += __shfl_xor(z[r], 1, 32);
            z[r] += __shfl_xor(z[r], 2, 32);
            z[r] += __shfl_xor(z[r], 4, 32);
            z[r] += __shfl_xor(z[r], 8, 32);
        }
        float s = 0.f;
        #pragma unroll
        for (int r = 0; r < 8; ++r) {
            float zz = z[r] + bP2v;
            s += fmaxf(zz, 0.f) + log1pf(__expf(-fabsf(zz)));  // stable softplus
        }
        s += __shfl_xor(s, 16, 32);   // combine the two half-wave totals
        wtotal += s;
    }
    if (lane == 0) wave_sums[wid] = wtotal;   // fixed slot -> deterministic
}

// Deterministic tree reduction + sqrt.
__global__ __launch_bounds__(256) void finalize_sum(
    const float* __restrict__ wsums, int nw, float* __restrict__ out)
{
    __shared__ float sm[256];
    float s = 0.f;
    for (int i = threadIdx.x; i < nw; i += 256) s += wsums[i];
    sm[threadIdx.x] = s;
    __syncthreads();
    for (int off = 128; off > 0; off >>= 1) {
        if (threadIdx.x < off) sm[threadIdx.x] += sm[threadIdx.x + off];
        __syncthreads();
    }
    if (threadIdx.x == 0) out[0] = sqrtf(sm[0]);
}

extern "C" void kernel_launch(void* const* d_in, const int* in_sizes, int n_in,
                              void* d_out, int out_size, void* d_ws, size_t ws_size,
                              hipStream_t stream)
{
    const float* node_feats = (const float*)d_in[0];
    const float* tOD = (const float*)d_in[1];
    const int*   Oi  = (const int*)d_in[2];
    const int*   Di  = (const int*)d_in[3];
    const float* WO1 = (const float*)d_in[4];  const float* bO1 = (const float*)d_in[5];
    const float* WO2 = (const float*)d_in[6];  const float* bO2 = (const float*)d_in[7];
    const float* WD1 = (const float*)d_in[8];  const float* bD1 = (const float*)d_in[9];
    const float* WD2 = (const float*)d_in[10]; const float* bD2 = (const float*)d_in[11];
    const float* WP1 = (const float*)d_in[12]; const float* bP1 = (const float*)d_in[13];
    const float* WP2 = (const float*)d_in[14]; const float* bP2 = (const float*)d_in[15];

    const int nNodes = in_sizes[0] / 64;         // 50000 (multiple of 16)
    const int nPairs = in_sizes[1];              // 2000000 (multiple of 16)
    const int nodeStrips = nNodes / 16;          // 3125
    const int pairStrips = nPairs / 16;          // 125000

    const int PAIR_BLOCKS = 512;
    const int NW = PAIR_BLOCKS * PAIR_WAVES;     // 4096 deterministic partial sums

    // Workspace layout (~6.7 MB total):
    char* ws = (char*)d_ws;
    float*        wave_sums = (float*)ws;                               // 16 KB
    _Float16*     Oe = (_Float16*)(ws + (1 << 16));                     // 3.2 MB
    _Float16*     De = (_Float16*)(ws + (1 << 16) + 3276800);           // 3.2 MB
    unsigned int* packs = (unsigned int*)(ws + (1 << 16) + 2 * 3276800);
    unsigned int* pO1 = packs;            // 16 tiles
    unsigned int* pO2 = pO1 + 16 * 256;   // 8 tiles
    unsigned int* pD1 = pO2 + 8 * 256;    // 16 tiles
    unsigned int* pD2 = pD1 + 16 * 256;   // 8 tiles
    unsigned int* pP1 = pD2 + 8 * 256;    // 8 tiles

    // 1) Pre-swizzle all B matrices into CDNA5 WMMA B-operand layout (f16).
    pack_b_tiles<<<16, 256, 0, stream>>>(WO1, 128, 2, 8, pO1);
    pack_b_tiles<<< 8, 256, 0, stream>>>(WO2,  32, 4, 2, pO2);
    pack_b_tiles<<<16, 256, 0, stream>>>(WD1, 128, 2, 8, pD1);
    pack_b_tiles<<< 8, 256, 0, stream>>>(WD2,  32, 4, 2, pD2);
    pack_b_tiles<<< 8, 256, 0, stream>>>(WP1,  64, 2, 4, pP1);   // rows 0..63 of [65x64]

    // 2) Encode all unique nodes (O and D weight sets via gridDim.y).
    dim3 ngrid((nodeStrips + NODE_WAVES - 1) / NODE_WAVES, 2, 1);
    node_encode<<<ngrid, 128, 0, stream>>>(node_feats,
                                           pO1, pO2, bO1, bO2, Oe,
                                           pD1, pD2, bD1, bD2, De,
                                           nodeStrips);

    // 3) Pair MLP + softplus, per-wave partial sums.
    pair_mlp<<<PAIR_BLOCKS, 256, 0, stream>>>(Oe, De, Oi, Di, tOD,
                                              pP1, bP1, WP1 + 64 * 64, WP2, bP2,
                                              wave_sums, pairStrips);

    // 4) Deterministic reduction + sqrt.
    finalize_sum<<<1, 256, 0, stream>>>(wave_sums, NW, (float*)d_out);
}
